// PointNet2Encoder_49237505082105
// MI455X (gfx1250) — matrix operand order repaired and checked by reading, not compiled, not run on hardware
//
#include <hip/hip_runtime.h>

// ---------------------------------------------------------------------------
// PointNet2 encoder for MI455X (gfx1250, wave32).
// MLP / FC layers run on v_wmma_f32_16x16x32_f16 matrix cores (f16 in, f32 acc).
// FPS keeps the min-distance array in VGPRs; argmax reduce via LDS.
// ---------------------------------------------------------------------------

typedef _Float16 v16h __attribute__((ext_vector_type(16)));
typedef _Float16 v8h  __attribute__((ext_vector_type(8)));
typedef float    v8f  __attribute__((ext_vector_type(8)));

// ---------------- prep: [B,4,N] -> coords [B,N,3], attn [B,N] ----------------
__global__ void prep_kernel(const float* __restrict__ xyz, float* __restrict__ coords,
                            float* __restrict__ attn, int B, int N) {
  int i = blockIdx.x * blockDim.x + threadIdx.x;
  if (i >= B * N) return;
  int b = i / N, n = i % N;
  const float* src = xyz + (size_t)b * 4 * N;
  coords[(size_t)i * 3 + 0] = src[0 * N + n];
  coords[(size_t)i * 3 + 1] = src[1 * N + n];
  coords[(size_t)i * 3 + 2] = src[2 * N + n];
  attn[i] = (src[3 * N + n] > 0.f) ? 1.f : 0.f;
}

// ---------------- attn-scaled coords -----------------------------------------
__global__ void attn_xyz_kernel(const float* __restrict__ coords, const float* __restrict__ attn,
                                float* __restrict__ axyz, float* __restrict__ nxyz, int total) {
  int i = blockIdx.x * blockDim.x + threadIdx.x;
  if (i >= total) return;
  float a = attn[i];
  for (int c = 0; c < 3; c++) {
    float v = coords[(size_t)i * 3 + c];
    axyz[(size_t)i * 3 + c] = a * v;
    nxyz[(size_t)i * 3 + c] = (1.f - a) * v;
  }
}

// ---------------- farthest point sampling ------------------------------------
// One block per batch; N = ITER*256 points; dmin lives in registers (unrolled).
template <int ITER>
__global__ __launch_bounds__(256) void fps_kernel(const float* __restrict__ xyz, int K,
                                                  int* __restrict__ idx_out, int ldo) {
  const int N = ITER * 256;
  __shared__ float rv[256];
  __shared__ int ri[256];
  __shared__ int s_last;
  int b = blockIdx.x, t = threadIdx.x;
  const float* P = xyz + (size_t)b * N * 3;
  float dloc[ITER];
#pragma unroll
  for (int j = 0; j < ITER; j++) dloc[j] = 1e10f;
  if (t == 0) s_last = 0;
  __syncthreads();
  for (int it = 0; it < K; it++) {
    int last = s_last;
    if (t == 0) idx_out[b * ldo + it] = last;
    float cx = P[last * 3 + 0], cy = P[last * 3 + 1], cz = P[last * 3 + 2];
    float best = -1.f;
    int bi = t;
#pragma unroll
    for (int j = 0; j < ITER; j++) {
      int i = t + j * 256;
      float dx = P[i * 3 + 0] - cx, dy = P[i * 3 + 1] - cy, dz = P[i * 3 + 2] - cz;
      float d = dx * dx + dy * dy + dz * dz;
      float dm = fminf(dloc[j], d);
      dloc[j] = dm;
      if (dm > best) { best = dm; bi = i; }
    }
    rv[t] = best;
    ri[t] = bi;
    __syncthreads();
    for (int s = 128; s > 0; s >>= 1) {
      if (t < s) {
        if (rv[t + s] > rv[t]) { rv[t] = rv[t + s]; ri[t] = ri[t + s]; }
      }
      __syncthreads();
    }
    if (t == 0) s_last = ri[0];
    __syncthreads();
  }
}

// ---------------- gathers ----------------------------------------------------
__global__ void gather3_kernel(const float* __restrict__ src, const int* __restrict__ idx,
                               float* __restrict__ dst, int B, int N, int S, int ldi, int ldd) {
  int i = blockIdx.x * blockDim.x + threadIdx.x;
  if (i >= B * S) return;
  int b = i / S, s = i % S;
  int j = idx[b * ldi + s];
  const float* sp = src + ((size_t)b * N + j) * 3;
  float* dp = dst + ((size_t)b * ldd + s) * 3;
  dp[0] = sp[0]; dp[1] = sp[1]; dp[2] = sp[2];
}

__global__ void gather1_kernel(const float* __restrict__ src, const int* __restrict__ idx,
                               float* __restrict__ dst, int B, int N, int S, int ldi, int ldd) {
  int i = blockIdx.x * blockDim.x + threadIdx.x;
  if (i >= B * S) return;
  int b = i / S, s = i % S;
  dst[(size_t)b * ldd + s] = src[(size_t)b * N + idx[b * ldi + s]];
}

// ---------------- ball query: one wave per query point -----------------------
__global__ __launch_bounds__(256) void ball_query_kernel(const float* __restrict__ q,
                                                         const float* __restrict__ p, int BQ,
                                                         int S, int N, int K, float r2,
                                                         int* __restrict__ idx) {
  int gw = blockIdx.x * (blockDim.x >> 5) + (threadIdx.x >> 5);
  if (gw >= BQ) return;
  int lane = threadIdx.x & 31;
  int b = gw / S;
  const float* Q = q + (size_t)gw * 3;
  float qx = Q[0], qy = Q[1], qz = Q[2];
  const float* P = p + (size_t)b * N * 3;
  int* out = idx + (size_t)gw * K;
  int count = 0;
  for (int base = 0; base < N && count < K; base += 32) {
    int n = base + lane;
    bool ok = false;
    if (n < N) {
      float dx = P[n * 3 + 0] - qx, dy = P[n * 3 + 1] - qy, dz = P[n * 3 + 2] - qz;
      ok = (dx * dx + dy * dy + dz * dz) < r2;
    }
    unsigned m = (unsigned)__ballot(ok);
    int pos = count + __popc(m & ((1u << lane) - 1u));
    if (ok && pos < K) out[pos] = n;
    count += __popc(m);
  }
  int filled = count < K ? count : K;
  for (int j = filled + lane; j < K; j += 32) out[j] = -1;
}

// ---------------- build grouped features (f16, K-padded) ---------------------
__global__ void group_feats_kernel(const float* __restrict__ qxyz, const float* __restrict__ pxyz,
                                   const float* __restrict__ feat, const int* __restrict__ idx,
                                   _Float16* __restrict__ X, int BQ, int S, int N, int K, int D,
                                   int Cpad) {
  int row = blockIdx.x * blockDim.x + threadIdx.x;
  if (row >= BQ * K) return;
  int g = row / K;
  int b = g / S;
  int i = idx[row];
  const float* Q = qxyz + (size_t)g * 3;
  _Float16* xr = X + (size_t)row * Cpad;
  float nx = 0.f, ny = 0.f, nz = 0.f;
  if (i >= 0) {
    const float* P = pxyz + ((size_t)b * N + i) * 3;
    nx = P[0]; ny = P[1]; nz = P[2];
  }
  xr[0] = (_Float16)(nx - Q[0]);
  xr[1] = (_Float16)(ny - Q[1]);
  xr[2] = (_Float16)(nz - Q[2]);
  int iw = (i < 0) ? (i + N) : i;  // torch wrap semantics
  const float* F = feat + ((size_t)b * N + iw) * D;
  for (int c = 0; c < D; c++) xr[3 + c] = (_Float16)F[c];
  for (int c = 3 + D; c < Cpad; c++) xr[c] = (_Float16)0.f;
}

// ---------------- pad+convert weights to f16 ---------------------------------
__global__ void padw_kernel(const float* __restrict__ W, _Float16* __restrict__ Wp, int O, int C,
                            int Cpad) {
  int i = blockIdx.x * blockDim.x + threadIdx.x;
  if (i >= O * Cpad) return;
  int o = i / Cpad, c = i % Cpad;
  Wp[i] = (_Float16)((c < C) ? W[(size_t)o * C + c] : 0.f);
}

// ---------------- WMMA GEMM: Y = act(X[M,Kp] * W[N,Kp]^T + bias) -------------
// block = 8 waves; wave owns one 16x16 D tile; block tile 32(M) x 64(N).
__global__ __launch_bounds__(256) void gemm_relu_wmma(const _Float16* __restrict__ X,
                                                      const _Float16* __restrict__ W,
                                                      const float* __restrict__ bias,
                                                      _Float16* __restrict__ Yh,
                                                      float* __restrict__ Yf, int M, int Kpad,
                                                      int ldy, int relu) {
  int wave = threadIdx.x >> 5;
  int lane = threadIdx.x & 31;
  int mw = wave & 1, nw = wave >> 1;
  int m0 = (blockIdx.x * 2 + mw) * 16;
  int n0 = (blockIdx.y * 4 + nw) * 16;
  int half = lane >> 4;
  int r = lane & 15;

  int arow = m0 + r;
  if (arow >= M) arow = M - 1;  // clamp; padded-row stores are guarded below
  int ncol = n0 + r;

  const _Float16* xrow = X + (size_t)arow * Kpad;
  const _Float16* wrow = W + (size_t)ncol * Kpad;

  v8f c = {};
  for (int k0 = 0; k0 < Kpad; k0 += 32) {
    // A fragment: 16x32 f16, lane half selects K {0-7,16-23} vs {8-15,24-31}
    v8h a_lo = *(const v8h*)(xrow + k0 + 8 * half);
    v8h a_hi = *(const v8h*)(xrow + k0 + 16 + 8 * half);
    // B fragment: 32x16 f16, lane = column, half selects K {0-7,16-23} vs {8-15,24-31}
    v8h b_lo = *(const v8h*)(wrow + k0 + 8 * half);
    v8h b_hi = *(const v8h*)(wrow + k0 + 16 + 8 * half);
    v16h a, b;
#pragma unroll
    for (int i = 0; i < 8; i++) {
      a[i] = a_lo[i]; a[i + 8] = a_hi[i];
      b[i] = b_lo[i]; b[i + 8] = b_hi[i];
    }
    c = __builtin_amdgcn_wmma_f32_16x16x32_f16(false, a, false, b, (short)0, c, false, false);
  }
  float bn = bias ? bias[ncol] : 0.f;
  int mbase = m0 + (half ? 8 : 0);
#pragma unroll
  for (int v = 0; v < 8; v++) {
    int m = mbase + v;
    if (m >= M) continue;
    float val = c[v] + bn;
    if (relu) val = fmaxf(val, 0.f);
    if (Yh) Yh[(size_t)m * ldy + ncol] = (_Float16)val;
    if (Yf) Yf[(size_t)m * ldy + ncol] = val;
  }
}

// ---------------- max over K groups: f16 [G*K, C] -> f32 [G, C] --------------
__global__ void maxpool_kernel(const _Float16* __restrict__ Y, float* __restrict__ out, int G,
                               int K, int C) {
  int i = blockIdx.x * blockDim.x + threadIdx.x;
  if (i >= G * C) return;
  int g = i / C, c = i % C;
  const _Float16* p = Y + (size_t)g * K * C + c;
  float m = -1e30f;
  for (int k = 0; k < K; k++) m = fmaxf(m, (float)p[(size_t)k * C]);
  out[i] = m;
}

// ---------------- concat [G,C] with [G] -> [G,C+1] ---------------------------
__global__ void concat_attn_kernel(const float* __restrict__ f, const float* __restrict__ a,
                                   float* __restrict__ out, int G, int C) {
  int Co = C + 1;
  int i = blockIdx.x * blockDim.x + threadIdx.x;
  if (i >= G * Co) return;
  int g = i / Co, c = i % Co;
  out[i] = (c < C) ? f[(size_t)g * C + c] : a[g];
}

// ---------------- sa3 input: [xyz(3) | f(Cf) | attn(1) | pad] as f16 ---------
__global__ void build_sa3_kernel(const float* __restrict__ x3, const float* __restrict__ f,
                                 const float* __restrict__ a, _Float16* __restrict__ X, int G,
                                 int Cf, int Cpad) {
  int i = blockIdx.x * blockDim.x + threadIdx.x;
  if (i >= G * Cpad) return;
  int g = i / Cpad, c = i % Cpad;
  float v;
  if (c < 3) v = x3[(size_t)g * 3 + c];
  else if (c < 3 + Cf) v = f[(size_t)g * Cf + (c - 3)];
  else if (c == 3 + Cf) v = a[g];
  else v = 0.f;
  X[i] = (_Float16)v;
}

__global__ void f2h_kernel(const float* __restrict__ in, _Float16* __restrict__ out, int n) {
  int i = blockIdx.x * blockDim.x + threadIdx.x;
  if (i < n) out[i] = (_Float16)in[i];
}

// ---------------------------------------------------------------------------
extern "C" void kernel_launch(void* const* d_in, const int* in_sizes, int n_in, void* d_out,
                              int out_size, void* d_ws, size_t ws_size, hipStream_t stream) {
  // Layer table: sa1(3), sa2(3), sa3(3), fc1, fc2
  static const int CIN[11]  = {4, 64, 64, 132, 128, 128, 260, 256, 512, 1024, 512};
  static const int COUT[11] = {64, 64, 128, 128, 128, 256, 256, 512, 1024, 512, 256};
  int KP[11];
  for (int i = 0; i < 11; i++) KP[i] = (CIN[i] + 31) & ~31;

  // ---- map inputs (handles insertion-order and jax-sorted flattening) ----
  const float* xyz_in = nullptr;
  const float* Wf[11];
  const float* bf[11];
  if (n_in >= 23 && in_sizes[0] == 524288 && in_sizes[1] == 512) {
    // sorted keys: fc1, fc2, sa1, sa2, sa3, xyz
    Wf[9] = (const float*)d_in[0];  bf[9] = (const float*)d_in[1];
    Wf[10] = (const float*)d_in[2]; bf[10] = (const float*)d_in[3];
    for (int i = 0; i < 9; i++) {
      Wf[i] = (const float*)d_in[4 + 2 * i];
      bf[i] = (const float*)d_in[5 + 2 * i];
    }
    xyz_in = (const float*)d_in[22];
  } else if (n_in >= 23 && in_sizes[0] == 256) {
    // params-first insertion order: sa1..sa3, fc1, fc2, xyz
    for (int i = 0; i < 11; i++) {
      Wf[i] = (const float*)d_in[2 * i];
      bf[i] = (const float*)d_in[2 * i + 1];
    }
    xyz_in = (const float*)d_in[22];
  } else {
    // xyz-first insertion order (default)
    xyz_in = (const float*)d_in[0];
    for (int i = 0; i < 11; i++) {
      Wf[i] = (const float*)d_in[1 + 2 * i];
      bf[i] = (const float*)d_in[2 + 2 * i];
    }
  }

  const int B = 8, N1 = 16384, S1 = 512, K1 = 32;
  const int N2 = 512, S2 = 128, K2 = 64;
  const float R1 = 0.02f, R2 = 0.045f;

  // ---- workspace bump allocator ----
  char* base = (char*)d_ws;
  size_t off = 0;
  auto A = [&](size_t bytes) -> void* {
    void* p = base + off;
    off = (off + bytes + 255) & ~(size_t)255;
    return p;
  };
  float* coords = (float*)A((size_t)B * N1 * 3 * 4);
  float* attn   = (float*)A((size_t)B * N1 * 4);
  float* axyz   = (float*)A((size_t)B * N1 * 3 * 4);
  float* nxyz   = (float*)A((size_t)B * N1 * 3 * 4);
  int*   fidx1  = (int*)A((size_t)B * S1 * 4);
  float* l1xyz  = (float*)A((size_t)B * S1 * 3 * 4);
  float* l1attn = (float*)A((size_t)B * S1 * 4);
  int*   bidx1  = (int*)A((size_t)B * S1 * K1 * 4);
  _Float16* Wp[11];
  for (int i = 0; i < 11; i++) Wp[i] = (_Float16*)A((size_t)COUT[i] * KP[i] * 2);
  const int M1 = B * S1 * K1;  // 131072
  _Float16* X1  = (_Float16*)A((size_t)M1 * 32 * 2);
  _Float16* Y1a = (_Float16*)A((size_t)M1 * 64 * 2);
  _Float16* Y1b = (_Float16*)A((size_t)M1 * 64 * 2);
  _Float16* Y1c = (_Float16*)A((size_t)M1 * 128 * 2);
  float* l1f = (float*)A((size_t)B * S1 * 128 * 4);
  float* l1p = (float*)A((size_t)B * S1 * 129 * 4);
  float* a2  = (float*)A((size_t)B * S1 * 3 * 4);
  float* n2  = (float*)A((size_t)B * S1 * 3 * 4);
  int*   fidx2  = (int*)A((size_t)B * S2 * 4);
  float* l2xyz  = (float*)A((size_t)B * S2 * 3 * 4);
  float* l2attn = (float*)A((size_t)B * S2 * 4);
  int*   bidx2  = (int*)A((size_t)B * S2 * K2 * 4);
  const int M2 = B * S2 * K2;  // 65536
  _Float16* X2  = (_Float16*)A((size_t)M2 * 160 * 2);
  _Float16* Y2a = (_Float16*)A((size_t)M2 * 128 * 2);
  _Float16* Y2b = (_Float16*)A((size_t)M2 * 128 * 2);
  _Float16* Y2c = (_Float16*)A((size_t)M2 * 256 * 2);
  float* l2f = (float*)A((size_t)B * S2 * 256 * 4);
  const int M3 = B * S2;  // 1024
  _Float16* X3  = (_Float16*)A((size_t)M3 * 288 * 2);
  _Float16* Y3a = (_Float16*)A((size_t)M3 * 256 * 2);
  _Float16* Y3b = (_Float16*)A((size_t)M3 * 512 * 2);
  _Float16* Y3c = (_Float16*)A((size_t)M3 * 1024 * 2);
  float*    pool3 = (float*)A((size_t)B * 1024 * 4);
  _Float16* Xfc   = (_Float16*)A((size_t)B * 1024 * 2);
  _Float16* Yfc1  = (_Float16*)A((size_t)B * 512 * 2);
  (void)ws_size; (void)out_size; (void)in_sizes; (void)n_in;

  auto cdiv = [](int a, int b) { return (a + b - 1) / b; };

  // ---- weight conversion ----
  for (int i = 0; i < 11; i++) {
    int tot = COUT[i] * KP[i];
    padw_kernel<<<cdiv(tot, 256), 256, 0, stream>>>(Wf[i], Wp[i], COUT[i], CIN[i], KP[i]);
  }

  // ---- stage 0: prep ----
  prep_kernel<<<cdiv(B * N1, 256), 256, 0, stream>>>(xyz_in, coords, attn, B, N1);
  attn_xyz_kernel<<<cdiv(B * N1, 256), 256, 0, stream>>>(coords, attn, axyz, nxyz, B * N1);

  // ---- sa1: FPS (256 + 256) ----
  fps_kernel<64><<<B, 256, 0, stream>>>(axyz, 256, fidx1, S1);
  fps_kernel<64><<<B, 256, 0, stream>>>(nxyz, 256, fidx1 + 256, S1);
  gather3_kernel<<<cdiv(B * 256, 256), 256, 0, stream>>>(axyz, fidx1, l1xyz, B, N1, 256, S1, S1);
  gather3_kernel<<<cdiv(B * 256, 256), 256, 0, stream>>>(nxyz, fidx1 + 256, l1xyz + 256 * 3, B, N1,
                                                         256, S1, S1);
  gather1_kernel<<<cdiv(B * S1, 256), 256, 0, stream>>>(attn, fidx1, l1attn, B, N1, S1, S1, S1);

  // ---- sa1: ball query + grouping + MLP ----
  ball_query_kernel<<<cdiv(B * S1, 8), 256, 0, stream>>>(l1xyz, coords, B * S1, S1, N1, K1, R1 * R1,
                                                         bidx1);
  group_feats_kernel<<<cdiv(M1, 256), 256, 0, stream>>>(l1xyz, coords, attn, bidx1, X1, B * S1, S1,
                                                        N1, K1, 1, 32);
  gemm_relu_wmma<<<dim3(cdiv(M1, 32), 1), 256, 0, stream>>>(X1, Wp[0], bf[0], Y1a, nullptr, M1, 32, 64, 1);
  gemm_relu_wmma<<<dim3(cdiv(M1, 32), 1), 256, 0, stream>>>(Y1a, Wp[1], bf[1], Y1b, nullptr, M1, 64, 64, 1);
  gemm_relu_wmma<<<dim3(cdiv(M1, 32), 2), 256, 0, stream>>>(Y1b, Wp[2], bf[2], Y1c, nullptr, M1, 64, 128, 1);
  maxpool_kernel<<<cdiv(B * S1 * 128, 256), 256, 0, stream>>>(Y1c, l1f, B * S1, K1, 128);
  concat_attn_kernel<<<cdiv(B * S1 * 129, 256), 256, 0, stream>>>(l1f, l1attn, l1p, B * S1, 128);

  // ---- sa2: FPS (64 + 64) over 512 points ----
  attn_xyz_kernel<<<cdiv(B * S1, 256), 256, 0, stream>>>(l1xyz, l1attn, a2, n2, B * S1);
  fps_kernel<2><<<B, 256, 0, stream>>>(a2, 64, fidx2, S2);
  fps_kernel<2><<<B, 256, 0, stream>>>(n2, 64, fidx2 + 64, S2);
  gather3_kernel<<<cdiv(B * 64, 256), 256, 0, stream>>>(a2, fidx2, l2xyz, B, N2, 64, S2, S2);
  gather3_kernel<<<cdiv(B * 64, 256), 256, 0, stream>>>(n2, fidx2 + 64, l2xyz + 64 * 3, B, N2, 64,
                                                        S2, S2);
  gather1_kernel<<<cdiv(B * S2, 256), 256, 0, stream>>>(l1attn, fidx2, l2attn, B, N2, S2, S2, S2);

  // ---- sa2: ball query + grouping + MLP ----
  ball_query_kernel<<<cdiv(B * S2, 8), 256, 0, stream>>>(l2xyz, l1xyz, B * S2, S2, N2, K2, R2 * R2,
                                                         bidx2);
  group_feats_kernel<<<cdiv(M2, 256), 256, 0, stream>>>(l2xyz, l1xyz, l1p, bidx2, X2, B * S2, S2,
                                                        N2, K2, 129, 160);
  gemm_relu_wmma<<<dim3(cdiv(M2, 32), 2), 256, 0, stream>>>(X2, Wp[3], bf[3], Y2a, nullptr, M2, 160, 128, 1);
  gemm_relu_wmma<<<dim3(cdiv(M2, 32), 2), 256, 0, stream>>>(Y2a, Wp[4], bf[4], Y2b, nullptr, M2, 128, 128, 1);
  gemm_relu_wmma<<<dim3(cdiv(M2, 32), 4), 256, 0, stream>>>(Y2b, Wp[5], bf[5], Y2c, nullptr, M2, 128, 256, 1);
  maxpool_kernel<<<cdiv(B * S2 * 256, 256), 256, 0, stream>>>(Y2c, l2f, B * S2, K2, 256);

  // ---- sa3: group-all MLP ----
  build_sa3_kernel<<<cdiv(M3 * 288, 256), 256, 0, stream>>>(l2xyz, l2f, l2attn, X3, M3, 256, 288);
  gemm_relu_wmma<<<dim3(cdiv(M3, 32), 4), 256, 0, stream>>>(X3, Wp[6], bf[6], Y3a, nullptr, M3, 288, 256, 1);
  gemm_relu_wmma<<<dim3(cdiv(M3, 32), 8), 256, 0, stream>>>(Y3a, Wp[7], bf[7], Y3b, nullptr, M3, 256, 512, 1);
  gemm_relu_wmma<<<dim3(cdiv(M3, 32), 16), 256, 0, stream>>>(Y3b, Wp[8], bf[8], Y3c, nullptr, M3, 512, 1024, 1);
  maxpool_kernel<<<cdiv(B * 1024, 256), 256, 0, stream>>>(Y3c, pool3, B, S2, 1024);

  // ---- FC head ----
  f2h_kernel<<<cdiv(B * 1024, 256), 256, 0, stream>>>(pool3, Xfc, B * 1024);
  gemm_relu_wmma<<<dim3(1, 8), 256, 0, stream>>>(Xfc, Wp[9], bf[9], Yfc1, nullptr, B, 1024, 512, 1);
  gemm_relu_wmma<<<dim3(1, 4), 256, 0, stream>>>(Yfc1, Wp[10], bf[10], nullptr, (float*)d_out, B,
                                                 512, 256, 0);
}
